// Hippocampus_73100343378368
// MI455X (gfx1250) — compile-verified, matrix operand order checked by reading, not run on hardware
//
#include <hip/hip_runtime.h>
#include <hip/hip_bf16.h>

typedef __attribute__((ext_vector_type(16))) _Float16 v16h;
typedef __attribute__((ext_vector_type(8)))  float    v8f;
typedef __attribute__((ext_vector_type(4)))  int      v4i;

#define AS1 __attribute__((address_space(1)))
#define AS3 __attribute__((address_space(3)))

#if defined(__gfx1250__) && __has_builtin(__builtin_amdgcn_global_load_async_to_lds_b128)
#define HAVE_ASYNC_LDS 1
#else
#define HAVE_ASYNC_LDS 0
#endif

#define BATCH 512
#define MLEN  1024
#define KEYD  128
#define VALD  128

__device__ __forceinline__ void wait_async_le4() {
#if HAVE_ASYNC_LDS
#if __has_builtin(__builtin_amdgcn_s_wait_asynccnt)
  __builtin_amdgcn_s_wait_asynccnt(4);
#else
  asm volatile("s_wait_asynccnt 4" ::: "memory");
#endif
#endif
}
__device__ __forceinline__ void wait_async_le0() {
#if HAVE_ASYNC_LDS
#if __has_builtin(__builtin_amdgcn_s_wait_asynccnt)
  __builtin_amdgcn_s_wait_asynccnt(0);
#else
  asm volatile("s_wait_asynccnt 0" ::: "memory");
#endif
#endif
}

// K index (even) of half-pair j for 16-bit A/B fragments (ISA 16-bit 16x32 layout):
// lanes 0-15 hold K {0..7, 16..23}; lanes 16-31 hold K {8..15, 24..31}
__device__ __forceinline__ int pairK(int j, bool hi) {
  int base = (j < 4) ? (2 * j) : (16 + 2 * (j - 4));
  return base + (hi ? 8 : 0);
}

// ---------------- Kernel A: mq = Linear(concat(Linear(state), latent)) ----------------
__global__ __launch_bounds__(128) void mq_kernel(
    const float* __restrict__ state, const float* __restrict__ latent,
    const float* __restrict__ W_state, const float* __restrict__ b_state,
    const float* __restrict__ W_query, const float* __restrict__ b_query,
    float* __restrict__ mq) {
  __shared__ float sh_in[128];
  __shared__ float sh_s[128];  // [ s(64) | latent(64) ]
  const int b = blockIdx.x, tid = threadIdx.x;
  sh_in[tid] = state[b * 128 + tid];
  __syncthreads();
  if (tid < 64) {
    float acc = b_state[tid];
    for (int i = 0; i < 128; ++i) acc += sh_in[i] * W_state[tid * 128 + i];
    sh_s[tid] = acc;
  } else {
    sh_s[tid] = latent[b * 64 + (tid - 64)];
  }
  __syncthreads();
  float acc = b_query[tid];
  for (int i = 0; i < 128; ++i) acc += sh_s[i] * W_query[tid * 128 + i];
  mq[b * 128 + tid] = acc;
}

// ---------------- Kernel B: multi-head kernel memory read (the 1 GB streamer) -------------
__global__ __launch_bounds__(256) void memread_kernel(
    const float* __restrict__ mq,
    const float* __restrict__ epi_keys, const float* __restrict__ epi_vals,
    const float* __restrict__ hebb_keys, const float* __restrict__ hebb_vals,
    float* __restrict__ ans /* [B,2,VALD] */) {
  __shared__ float s_scores[4 * MLEN];                    // 16 KB: scores -> softmax weights
  __shared__ __align__(16) float s_stage[2 * 32 * VALD];  // 2 x 16 KB ping-pong vals staging
  __shared__ float s_red[4 * 64];
  __shared__ float s_bc[8];

  const int b = blockIdx.x;
  const int bank = blockIdx.y;
  const float* keys = (bank ? hebb_keys : epi_keys) + (size_t)b * MLEN * KEYD;
  const float* vals = (bank ? hebb_vals : epi_vals) + (size_t)b * MLEN * VALD;
  const int tid  = threadIdx.x;
  const int wid  = tid >> 5;
  const int lane = tid & 31;
  const bool hi  = lane >= 16;
  const int ln   = lane & 15;

  // ---- Build block-diagonal q fragments: B_s[k][h] = q[32s+k]*scale iff h==s
  const float qscale = 0.17677669529663687f;  // 1/sqrt(32)
  const float* mqb = mq + (size_t)b * KEYD;
  v16h Bq[4];
  for (int s = 0; s < 4; ++s) {
    for (int e = 0; e < 16; ++e) Bq[s][e] = (_Float16)0.f;
    if (ln == s) {
      for (int j = 0; j < 8; ++j) {
        int k = pairK(j, hi);
        Bq[s][2 * j]     = (_Float16)(mqb[32 * s + k] * qscale);
        Bq[s][2 * j + 1] = (_Float16)(mqb[32 * s + k + 1] * qscale);
      }
    }
  }

  // ---- Pass 1: scores[h][m] = (q_h . keys[m]_h) * scale via WMMA, D[m-tile, head]
  for (int t = wid; t < MLEN / 16; t += 8) {
    const int m0 = t * 16;
    v8f D;
    for (int e = 0; e < 8; ++e) D[e] = 0.f;
    const float* krow = keys + (size_t)(m0 + ln) * KEYD;
    if (t + 8 < MLEN / 16)
      __builtin_prefetch(keys + (size_t)(m0 + 128 + ln) * KEYD, 0, 0);  // global_prefetch_b8
    for (int s = 0; s < 4; ++s) {
      v16h A;
      for (int j = 0; j < 8; ++j) {
        int kd = 32 * s + pairK(j, hi);
        float2 p = *(const float2*)(krow + kd);
        A[2 * j]     = (_Float16)p.x;
        A[2 * j + 1] = (_Float16)p.y;
      }
      D = __builtin_amdgcn_wmma_f32_16x16x32_f16(false, A, false, Bq[s], (short)0, D,
                                                 false, false);
    }
    if (ln < 4) {  // column ln == head; lanes 0-15 rows m0+r, lanes 16-31 rows m0+8+r
      int mb = m0 + (hi ? 8 : 0);
      for (int r = 0; r < 8; ++r) s_scores[ln * MLEN + mb + r] = D[r];
    }
  }
  __syncthreads();

  // ---- Softmax over M per head (scores -> normalized weights, in LDS)
  {
    const int h = tid >> 6, t = tid & 63;
    float mx = -3.402823466e38f;
    for (int i = t; i < MLEN; i += 64) mx = fmaxf(mx, s_scores[h * MLEN + i]);
    s_red[h * 64 + t] = mx;
    __syncthreads();
    if (t == 0) {
      float m2 = s_red[h * 64];
      for (int i = 1; i < 64; ++i) m2 = fmaxf(m2, s_red[h * 64 + i]);
      s_bc[h] = m2;
    }
    __syncthreads();
    const float hm = s_bc[h];
    float psum = 0.f;
    for (int i = t; i < MLEN; i += 64) {
      float e = __expf(s_scores[h * MLEN + i] - hm);
      s_scores[h * MLEN + i] = e;
      psum += e;
    }
    s_red[h * 64 + t] = psum;
    __syncthreads();
    if (t == 0) {
      float s2 = 0.f;
      for (int i = 0; i < 64; ++i) s2 += s_red[h * 64 + i];
      s_bc[4 + h] = 1.f / s2;
    }
    __syncthreads();
    const float inv = s_bc[4 + h];
    for (int i = t; i < MLEN; i += 64) s_scores[h * MLEN + i] *= inv;
  }
  __syncthreads();

  // ---- Pass 2: out[c] = sum_m w[head(c)][m] * vals[m][c]; wave owns a 16-col tile.
  // Vals stream through a double-buffered LDS stage filled by async global->LDS DMA;
  // ASYNCcnt completes in order per wave, so after issuing the next block's 4 loads,
  // s_wait_asynccnt 4 guarantees the current block is resident.
  const int c0 = wid * 16;
  const int hc = c0 >> 5;  // head of this column tile (16-col tile lies in one head)
  v8f D2;
  for (int e = 0; e < 8; ++e) D2[e] = 0.f;

#if HAVE_ASYNC_LDS
  for (int i = 0; i < 4; ++i) {  // preload block 0 into buffer 0
    int f4 = i * 256 + tid;
    __builtin_amdgcn_global_load_async_to_lds_b128(
        (AS1 v4i*)(vals + (size_t)f4 * 4), (AS3 v4i*)&s_stage[f4 * 4], 0, 0);
  }
#endif
  for (int blk = 0; blk < MLEN / 32; ++blk) {
    const int mb = blk * 32;
    const int cur = (blk & 1) * (32 * VALD);
#if HAVE_ASYNC_LDS
    if (blk + 1 < MLEN / 32) {
      const int nxt = ((blk + 1) & 1) * (32 * VALD);
      for (int i = 0; i < 4; ++i) {
        int f4 = i * 256 + tid;
        __builtin_amdgcn_global_load_async_to_lds_b128(
            (AS1 v4i*)(vals + (size_t)(mb + 32) * VALD + (size_t)f4 * 4),
            (AS3 v4i*)&s_stage[nxt + f4 * 4], 0, 0);
      }
      wait_async_le4();
    } else {
      wait_async_le0();
    }
#else
    for (int i = 0; i < 4; ++i) {
      int f4 = i * 256 + tid;
      *(float4*)&s_stage[cur + f4 * 4] =
          *(const float4*)(vals + (size_t)mb * VALD + (size_t)f4 * 4);
    }
#endif
    __syncthreads();
    // A: rows 0-3 = softmax weights (others zero; their D rows are ignored)
    v16h A;
    for (int e = 0; e < 16; ++e) A[e] = (_Float16)0.f;
    if (ln < 4) {
      for (int j = 0; j < 8; ++j) {
        int k = pairK(j, hi);
        A[2 * j]     = (_Float16)s_scores[ln * MLEN + mb + k];
        A[2 * j + 1] = (_Float16)s_scores[ln * MLEN + mb + k + 1];
      }
    }
    // B: vals tile, col n = c0+ln, K pattern per 16-bit layout
    v16h Bv;
    for (int j = 0; j < 8; ++j) {
      int k = pairK(j, hi);
      Bv[2 * j]     = (_Float16)s_stage[cur + k * VALD + c0 + ln];
      Bv[2 * j + 1] = (_Float16)s_stage[cur + (k + 1) * VALD + c0 + ln];
    }
    D2 = __builtin_amdgcn_wmma_f32_16x16x32_f16(false, A, false, Bv, (short)0, D2,
                                                false, false);
    __syncthreads();  // all waves done reading buffer `cur` before it is refilled
  }
  // D row hc is held by lanes 0-15 in VGPR hc (hc < 4)
  if (!hi) ans[(size_t)b * 256 + bank * 128 + c0 + ln] = D2[hc];
}

// ---------------- Kernel C: combiner MHA (q_len=1, kv_len=2) + output proj ----------------
__global__ __launch_bounds__(128) void combine_kernel(
    const float* __restrict__ ans, const float* __restrict__ rim,
    const float* __restrict__ W_hq, const float* __restrict__ b_hq,
    const float* __restrict__ W_mk, const float* __restrict__ b_mk,
    const float* __restrict__ W_mv, const float* __restrict__ b_mv,
    const float* __restrict__ Wq, const float* __restrict__ Wk,
    const float* __restrict__ Wv, const float* __restrict__ Wo,
    float* __restrict__ out) {
  __shared__ float sh_ans[2][128], sh_rim[256];
  __shared__ float sh_q[128], sh_k[2][128], sh_v[2][128];
  __shared__ float sh_qp[128], sh_kp[2][128], sh_vp[2][128];
  __shared__ float sh_ctx[128];
  const int b = blockIdx.x, tid = threadIdx.x;
  sh_ans[0][tid] = ans[(size_t)b * 256 + tid];
  sh_ans[1][tid] = ans[(size_t)b * 256 + 128 + tid];
  sh_rim[tid] = rim[(size_t)b * 256 + tid];
  sh_rim[128 + tid] = rim[(size_t)b * 256 + 128 + tid];
  __syncthreads();
  {
    float acc = b_hq[tid];
    for (int i = 0; i < 256; ++i) acc += sh_rim[i] * W_hq[tid * 256 + i];
    sh_q[tid] = acc;
  }
  for (int m = 0; m < 2; ++m) {
    float ak = b_mk[tid], av = b_mv[tid];
    for (int i = 0; i < 128; ++i) {
      float x = sh_ans[m][i];
      ak += x * W_mk[tid * 128 + i];
      av += x * W_mv[tid * 128 + i];
    }
    sh_k[m][tid] = ak;
    sh_v[m][tid] = av;
  }
  __syncthreads();
  {
    float a = 0.f;
    for (int i = 0; i < 128; ++i) a += sh_q[i] * Wq[tid * 128 + i];
    sh_qp[tid] = a;
  }
  for (int m = 0; m < 2; ++m) {
    float a = 0.f, c = 0.f;
    for (int i = 0; i < 128; ++i) {
      a += sh_k[m][i] * Wk[tid * 128 + i];
      c += sh_v[m][i] * Wv[tid * 128 + i];
    }
    sh_kp[m][tid] = a;
    sh_vp[m][tid] = c;
  }
  __syncthreads();
  {
    const int h = tid >> 5;
    float s0 = 0.f, s1 = 0.f;
    for (int d = 0; d < 32; ++d) {
      float qv = sh_qp[h * 32 + d];
      s0 += qv * sh_kp[0][h * 32 + d];
      s1 += qv * sh_kp[1][h * 32 + d];
    }
    const float sc = 0.17677669529663687f;  // 1/sqrt(32)
    s0 *= sc;
    s1 *= sc;
    float mx = fmaxf(s0, s1);
    float e0 = __expf(s0 - mx), e1 = __expf(s1 - mx);
    float inv = 1.f / (e0 + e1);
    sh_ctx[tid] = (e0 * sh_vp[0][tid] + e1 * sh_vp[1][tid]) * inv;
  }
  __syncthreads();
  {
    float a = 0.f;
    for (int i = 0; i < 128; ++i) a += sh_ctx[i] * Wo[tid * 128 + i];
    out[(size_t)b * 128 + tid] = a;
  }
}

extern "C" void kernel_launch(void* const* d_in, const int* in_sizes, int n_in,
                              void* d_out, int out_size, void* d_ws, size_t ws_size,
                              hipStream_t stream) {
  (void)in_sizes; (void)n_in; (void)out_size; (void)ws_size;
  const float* state     = (const float*)d_in[0];
  const float* latent    = (const float*)d_in[1];
  const float* rim       = (const float*)d_in[2];
  const float* epi_keys  = (const float*)d_in[3];
  const float* epi_vals  = (const float*)d_in[4];
  const float* hebb_keys = (const float*)d_in[5];
  const float* hebb_vals = (const float*)d_in[6];
  const float* W_state = (const float*)d_in[7];
  const float* b_state = (const float*)d_in[8];
  const float* W_query = (const float*)d_in[9];
  const float* b_query = (const float*)d_in[10];
  const float* W_hq = (const float*)d_in[11];
  const float* b_hq = (const float*)d_in[12];
  const float* W_mk = (const float*)d_in[13];
  const float* b_mk = (const float*)d_in[14];
  const float* W_mv = (const float*)d_in[15];
  const float* b_mv = (const float*)d_in[16];
  const float* Wq = (const float*)d_in[17];
  const float* Wk = (const float*)d_in[18];
  const float* Wv = (const float*)d_in[19];
  const float* Wo = (const float*)d_in[20];
  float* out = (float*)d_out;

  float* ws_mq  = (float*)d_ws;              // [512,128]
  float* ws_ans = ws_mq + BATCH * 128;       // [512,2,128]

  mq_kernel<<<BATCH, 128, 0, stream>>>(state, latent, W_state, b_state, W_query, b_query,
                                       ws_mq);
  memread_kernel<<<dim3(BATCH, 2), 256, 0, stream>>>(ws_mq, epi_keys, epi_vals, hebb_keys,
                                                     hebb_vals, ws_ans);
  combine_kernel<<<BATCH, 128, 0, stream>>>(ws_ans, rim, W_hq, b_hq, W_mk, b_mk, W_mv,
                                            b_mv, Wq, Wk, Wv, Wo, out);
}
